// DbrxExpertGLU_49228915147014
// MI455X (gfx1250) — matrix operand order repaired and checked by reading, not compiled, not run on hardware
//
#include <hip/hip_runtime.h>
#include <hip/hip_bf16.h>

// ---------------------------------------------------------------------------
// DBRX expert SwiGLU MLP for gfx1250 (MI455X), bf16 WMMA path.
//   gate = x @ W1^T ; up = x @ V1^T ; inter = silu(gate)*up  (bf16 -> d_ws)
//   down = inter @ W2                                        (f32  -> d_out)
//
// Pipeline per K-step (fenced with sched_barrier so hipcc cannot collapse the
// staging registers and serialize the loads with s_wait_loadcnt 0):
//   [issue global b128 loads] || [ds_load_b128 frags + v_wmma] || [cvt+ds_store]
// ---------------------------------------------------------------------------

typedef __attribute__((ext_vector_type(16))) __bf16 v16bf;
typedef __attribute__((ext_vector_type(8)))  __bf16 v8bf;
typedef __attribute__((ext_vector_type(4)))  __bf16 v4bf;
typedef __attribute__((ext_vector_type(8)))  float  v8f;

#define TOKENS 4096
#define HIDDEN 4096
#define FFN    14336

#define BM 64     // token rows per block
#define BN 128    // output cols per block
#define BK 32     // K depth per stage (one WMMA K)
#define LDK 40    // padded LDS row stride (bf16 elems) -> 80 B, 16B-aligned rows

union Frag { v16bf v; v8bf h[2]; };

__device__ __forceinline__ v4bf cvt4(float4 f) {
  v4bf r;
  r[0] = (__bf16)f.x; r[1] = (__bf16)f.y; r[2] = (__bf16)f.z; r[3] = (__bf16)f.w;
  return r;
}

// A-matrix fragment (16x32 bf16, MxK). ISA layout:
//   lanes 0-15 : K = 0..7 (VGPR0-3), 16..23 (VGPR4-7)
//   lanes16-31 : K = 8..15, 24..31
__device__ __forceinline__ v16bf load_frag_a(const __bf16* row, int half) {
  Frag f;
  const int c = half * 8;
  f.h[0] = *(const v8bf*)(row + c);
  f.h[1] = *(const v8bf*)(row + c + 16);
  return f.v;
}

// B-matrix fragment (32x16 bf16, KxN), per-lane column of B:
//   lanes 0-15 : K = 0..7, 16..23; lanes 16-31 : K = 8..15, 24..31
__device__ __forceinline__ v16bf load_frag_b(const __bf16* col, int half) {
  Frag f;
  const int c = half * 8;
  f.h[0] = *(const v8bf*)(col + c);
  f.h[1] = *(const v8bf*)(col + c + 16);
  return f.v;
}

// ---------------------------------------------------------------------------
// Kernel 1: fused gate/up GEMM + SwiGLU.
// Grid: (TOKENS/BM, FFN/BN) -- token dim fastest so consecutive blocks share
// the same W1/V1 tile in L2 (weights stream from HBM exactly once).
// ---------------------------------------------------------------------------
__global__ __launch_bounds__(256)
void swiglu_gateup_kernel(const float* __restrict__ x,
                          const float* __restrict__ w1,
                          const float* __restrict__ v1,
                          __bf16* __restrict__ inter) {
  __shared__ __bf16 As [2][BM][LDK];
  __shared__ __bf16 B1s[2][BN][LDK];
  __shared__ __bf16 B2s[2][BN][LDK];

  const int t     = threadIdx.x;
  const int lane  = t & 31;
  const int wave  = t >> 5;
  const int waveM = wave >> 2;   // 0..1
  const int waveN = wave & 3;    // 0..3
  const int half  = lane >> 4;
  const int l16   = lane & 15;

  const unsigned row0 = blockIdx.x * BM;   // token tile (fast-varying)
  const unsigned col0 = blockIdx.y * BN;   // ffn tile

  // per-thread staging coordinates (32-bit offsets -> saddr+voffset loads)
  const unsigned ar[2] = { (unsigned)(t >> 3), (unsigned)((t + 256) >> 3) };
  const unsigned ak[2] = { (unsigned)((t & 7) * 4), (unsigned)(((t + 256) & 7) * 4) };
  unsigned aOff[2], bOff[4];
  unsigned br[4], bk[4];
  for (int i = 0; i < 2; ++i) aOff[i] = (row0 + ar[i]) * HIDDEN + ak[i];
  for (int i = 0; i < 4; ++i) {
    const int idx = t + i * 256;
    br[i] = idx >> 3; bk[i] = (idx & 7) * 4;
    bOff[i] = (col0 + br[i]) * HIDDEN + bk[i];
  }

  v8f cg[2][2] = {};
  v8f cu[2][2] = {};

  float4 aReg[2], b1Reg[4], b2Reg[4];

  // ---- preload K-tile 0 into LDS buffer 0 ----
  for (int i = 0; i < 2; ++i)
    *(v4bf*)&As[0][ar[i]][ak[i]] = cvt4(*(const float4*)(x + aOff[i]));
  for (int i = 0; i < 4; ++i) {
    *(v4bf*)&B1s[0][br[i]][bk[i]] = cvt4(*(const float4*)(w1 + bOff[i]));
    *(v4bf*)&B2s[0][br[i]][bk[i]] = cvt4(*(const float4*)(v1 + bOff[i]));
  }
  __syncthreads();

  const int KT = HIDDEN / BK;   // 128
  for (int kt = 0; kt < KT; ++kt) {
    const int buf = kt & 1;

    // ---- phase 1: batch-issue next tile's global loads ----
    if (kt + 1 < KT) {
      const unsigned k0 = (unsigned)(kt + 1) * BK;
      for (int i = 0; i < 2; ++i) aReg[i]  = *(const float4*)(x  + aOff[i] + k0);
      for (int i = 0; i < 4; ++i) b1Reg[i] = *(const float4*)(w1 + bOff[i] + k0);
      for (int i = 0; i < 4; ++i) b2Reg[i] = *(const float4*)(v1 + bOff[i] + k0);
    }
    if (kt + 2 < KT) {  // warm GL2 for K+2 weight tiles
      const unsigned k2 = (unsigned)(kt + 2) * BK;
      __builtin_prefetch(w1 + bOff[0] + k2, 0, 0);
      __builtin_prefetch(v1 + bOff[0] + k2, 0, 0);
    }
    __builtin_amdgcn_sched_barrier(0);

    // ---- phase 2: fragments + 8 WMMAs per wave ----
    v16bf a[2];
    for (int mi = 0; mi < 2; ++mi)
      a[mi] = load_frag_a(&As[buf][waveM * 32 + mi * 16 + l16][0], half);

    for (int ni = 0; ni < 2; ++ni) {
      const v16bf b1 = load_frag_b(&B1s[buf][waveN * 32 + ni * 16 + l16][0], half);
      const v16bf b2 = load_frag_b(&B2s[buf][waveN * 32 + ni * 16 + l16][0], half);
      for (int mi = 0; mi < 2; ++mi) {
        cg[mi][ni] = __builtin_amdgcn_wmma_f32_16x16x32_bf16(
            false, a[mi], false, b1, (short)0, cg[mi][ni], false, false);
        cu[mi][ni] = __builtin_amdgcn_wmma_f32_16x16x32_bf16(
            false, a[mi], false, b2, (short)0, cu[mi][ni], false, false);
      }
    }
    __builtin_amdgcn_sched_barrier(0);

    // ---- phase 3: convert + LDS store (loads have had phase 2 to land) ----
    if (kt + 1 < KT) {
      const int nbuf = buf ^ 1;
      for (int i = 0; i < 2; ++i)
        *(v4bf*)&As[nbuf][ar[i]][ak[i]] = cvt4(aReg[i]);
      for (int i = 0; i < 4; ++i) {
        *(v4bf*)&B1s[nbuf][br[i]][bk[i]] = cvt4(b1Reg[i]);
        *(v4bf*)&B2s[nbuf][br[i]][bk[i]] = cvt4(b2Reg[i]);
      }
    }
    __syncthreads();
  }

  // epilogue: inter = silu(gate) * up, bf16 store.
  // C/D layout: VGPR r, lanes 0-15 -> M=r, lanes 16-31 -> M=r+8; N = lane&15.
  for (int mi = 0; mi < 2; ++mi)
    for (int ni = 0; ni < 2; ++ni)
      for (int r = 0; r < 8; ++r) {
        const float g = cg[mi][ni][r];
        const float u = cu[mi][ni][r];
        const float s = g / (1.0f + __expf(-g));
        const unsigned row = row0 + waveM * 32 + mi * 16 + half * 8 + r;
        const unsigned col = col0 + waveN * 32 + ni * 16 + l16;
        inter[(size_t)row * FFN + col] = (__bf16)(s * u);
      }
}

// ---------------------------------------------------------------------------
// Kernel 2: down = inter @ W2.  Grid: (TOKENS/BM, HIDDEN/BN), 256 threads.
// K = FFN (448 stages). W2 tile transposed via 4x4 in-register transpose ->
// ds_store_b64 (instead of 16 scattered b16 stores).
// ---------------------------------------------------------------------------
__global__ __launch_bounds__(256)
void down_proj_kernel(const __bf16* __restrict__ inter,
                      const float* __restrict__ w2,
                      float* __restrict__ out) {
  __shared__ __bf16 As[2][BM][LDK];
  __shared__ __bf16 Bs[2][BN][LDK];

  const int t     = threadIdx.x;
  const int lane  = t & 31;
  const int wave  = t >> 5;
  const int waveM = wave >> 2;
  const int waveN = wave & 3;
  const int half  = lane >> 4;
  const int l16   = lane & 15;

  const unsigned row0 = blockIdx.x * BM;   // token tile (fast-varying)
  const unsigned col0 = blockIdx.y * BN;   // hidden tile

  // A staging: one v8bf per thread
  const unsigned arow = t >> 2;            // 0..63
  const unsigned ak   = (t & 3) * 8;       // k chunk of 8 bf16
  const unsigned aOff = (row0 + arow) * FFN + ak;

  // B staging: 4 consecutive k-rows at one n-quad per thread
  const unsigned nq = (t & 31) * 4;        // col quad 0..124
  const unsigned kq = (t >> 5) * 4;        // k base 0..28
  const unsigned bOff = kq * HIDDEN + col0 + nq;   // row k=kq, col col0+nq

  v8f acc[2][2] = {};

  v8bf   aReg;
  float4 bReg[4];

  // ---- preload K-tile 0 ----
  {
    aReg = *(const v8bf*)(inter + aOff);
    *(v8bf*)&As[0][arow][ak] = aReg;
    for (int j = 0; j < 4; ++j)
      bReg[j] = *(const float4*)(w2 + bOff + (unsigned)j * HIDDEN);
    v4bf c0, c1, c2, c3;
    c0[0]=(__bf16)bReg[0].x; c0[1]=(__bf16)bReg[1].x; c0[2]=(__bf16)bReg[2].x; c0[3]=(__bf16)bReg[3].x;
    c1[0]=(__bf16)bReg[0].y; c1[1]=(__bf16)bReg[1].y; c1[2]=(__bf16)bReg[2].y; c1[3]=(__bf16)bReg[3].y;
    c2[0]=(__bf16)bReg[0].z; c2[1]=(__bf16)bReg[1].z; c2[2]=(__bf16)bReg[2].z; c2[3]=(__bf16)bReg[3].z;
    c3[0]=(__bf16)bReg[0].w; c3[1]=(__bf16)bReg[1].w; c3[2]=(__bf16)bReg[2].w; c3[3]=(__bf16)bReg[3].w;
    *(v4bf*)&Bs[0][nq + 0][kq] = c0;
    *(v4bf*)&Bs[0][nq + 1][kq] = c1;
    *(v4bf*)&Bs[0][nq + 2][kq] = c2;
    *(v4bf*)&Bs[0][nq + 3][kq] = c3;
  }
  __syncthreads();

  const int KT = FFN / BK;   // 448
  for (int kt = 0; kt < KT; ++kt) {
    const int buf = kt & 1;

    // ---- phase 1: batch-issue next tile's global loads ----
    if (kt + 1 < KT) {
      const unsigned k0 = (unsigned)(kt + 1) * BK;
      aReg = *(const v8bf*)(inter + aOff + k0);
      for (int j = 0; j < 4; ++j)
        bReg[j] = *(const float4*)(w2 + bOff + (k0 + (unsigned)j) * HIDDEN);
    }
    if (kt + 2 < KT) {
      const unsigned k2 = (unsigned)(kt + 2) * BK;
      __builtin_prefetch(w2 + bOff + k2 * HIDDEN, 0, 0);
    }
    __builtin_amdgcn_sched_barrier(0);

    // ---- phase 2: fragments + 4 WMMAs per wave ----
    v16bf a[2];
    for (int mi = 0; mi < 2; ++mi)
      a[mi] = load_frag_a(&As[buf][waveM * 32 + mi * 16 + l16][0], half);

    for (int ni = 0; ni < 2; ++ni) {
      const v16bf b = load_frag_b(&Bs[buf][waveN * 32 + ni * 16 + l16][0], half);
      for (int mi = 0; mi < 2; ++mi)
        acc[mi][ni] = __builtin_amdgcn_wmma_f32_16x16x32_bf16(
            false, a[mi], false, b, (short)0, acc[mi][ni], false, false);
    }
    __builtin_amdgcn_sched_barrier(0);

    // ---- phase 3: convert + transpose + LDS store ----
    if (kt + 1 < KT) {
      const int nbuf = buf ^ 1;
      *(v8bf*)&As[nbuf][arow][ak] = aReg;
      v4bf c0, c1, c2, c3;
      c0[0]=(__bf16)bReg[0].x; c0[1]=(__bf16)bReg[1].x; c0[2]=(__bf16)bReg[2].x; c0[3]=(__bf16)bReg[3].x;
      c1[0]=(__bf16)bReg[0].y; c1[1]=(__bf16)bReg[1].y; c1[2]=(__bf16)bReg[2].y; c1[3]=(__bf16)bReg[3].y;
      c2[0]=(__bf16)bReg[0].z; c2[1]=(__bf16)bReg[1].z; c2[2]=(__bf16)bReg[2].z; c2[3]=(__bf16)bReg[3].z;
      c3[0]=(__bf16)bReg[0].w; c3[1]=(__bf16)bReg[1].w; c3[2]=(__bf16)bReg[2].w; c3[3]=(__bf16)bReg[3].w;
      *(v4bf*)&Bs[nbuf][nq + 0][kq] = c0;
      *(v4bf*)&Bs[nbuf][nq + 1][kq] = c1;
      *(v4bf*)&Bs[nbuf][nq + 2][kq] = c2;
      *(v4bf*)&Bs[nbuf][nq + 3][kq] = c3;
    }
    __syncthreads();
  }

  for (int mi = 0; mi < 2; ++mi)
    for (int ni = 0; ni < 2; ++ni)
      for (int r = 0; r < 8; ++r) {
        const unsigned row = row0 + waveM * 32 + mi * 16 + half * 8 + r;
        const unsigned col = col0 + waveN * 32 + ni * 16 + l16;
        __builtin_nontemporal_store(acc[mi][ni][r], out + (size_t)row * HIDDEN + col);
      }
}

// ---------------------------------------------------------------------------
extern "C" void kernel_launch(void* const* d_in, const int* in_sizes, int n_in,
                              void* d_out, int out_size, void* d_ws, size_t ws_size,
                              hipStream_t stream) {
  const float* x  = (const float*)d_in[0];
  const float* w1 = (const float*)d_in[1];
  const float* v1 = (const float*)d_in[2];
  const float* w2 = (const float*)d_in[3];

  __bf16* inter = (__bf16*)d_ws;           // TOKENS*FFN bf16 = ~117 MB scratch
  float*  out   = (float*)d_out;

  dim3 blk(256);
  dim3 gridA(TOKENS / BM, FFN / BN);       // 64 x 112 (token dim fastest)
  hipLaunchKernelGGL(swiglu_gateup_kernel, gridA, blk, 0, stream, x, w1, v1, inter);

  dim3 gridB(TOKENS / BM, HIDDEN / BN);    // 64 x 32
  hipLaunchKernelGGL(down_proj_kernel, gridB, blk, 0, stream, inter, w2, out);
}
